// KumaSelfAttention_10917806866980
// MI455X (gfx1250) — matrix operand order repaired and checked by reading, not compiled, not run on hardware
//
#include <hip/hip_runtime.h>
#include <hip/hip_bf16.h>

// ---------------------------------------------------------------------------
// KumaSelfAttention for MI455X (gfx1250, wave32, WMMA f32_16x16x32_f16)
//
// Pipeline:
//   1) cvt q,k -> f16 ; transpose-cvt W* -> f16 [N,K]
//   2) 8x gemm_bias_relu_f16 (two-layer MLP for q_a, k_a, q_b, k_b)
//      - double-buffered LDS; async global->LDS copies (ASYNCcnt) when the
//        toolchain exposes them, else register-staged sync pipeline
//   3) kuma_attention: fused (q_a@k_a^T, q_b@k_b^T) + rel-dist + softplus
//      + HardKuma CDF/mean epilogue -> att [B,T,T] f32
// ---------------------------------------------------------------------------

typedef __attribute__((ext_vector_type(16))) _Float16 v16h;
typedef __attribute__((ext_vector_type(8)))  _Float16 v8h;
typedef __attribute__((ext_vector_type(8)))  float    v8f;

#define LDT    56   // LDS row stride in halfs: 112B = 7*16B (b128-aligned, conflict-free gathers)
#define LDC    136  // C-staging row stride in halfs: 272B = 17*16B
#define TILE_K 32
#define ABUF   (128 * LDT)   // halfs per tile buffer

#ifndef __has_builtin
#define __has_builtin(x) 0
#endif
#if __has_builtin(__builtin_amdgcn_global_load_async_to_lds_b128) && \
    __has_builtin(__builtin_amdgcn_s_wait_asynccnt)
#define ASYNC_OK 1
#else
#define ASYNC_OK 0
#endif

#if ASYNC_OK
// Builtin signature (from hipcc diagnostic): first param is
// '__device__ int __attribute__((vector_size(16)))*' -> AS(1) int4*;
// LDS side is the AS(3) int4*.
typedef int v4i_b128 __attribute__((vector_size(16)));
typedef __attribute__((address_space(1))) v4i_b128* gbl_b128_p;
typedef __attribute__((address_space(3))) v4i_b128* lds_b128_p;

// per-lane async copy of 16 bytes global -> LDS, tracked by ASYNCcnt
__device__ __forceinline__ void async_cp16(const _Float16* g, _Float16* l) {
  __builtin_amdgcn_global_load_async_to_lds_b128(
      (gbl_b128_p)const_cast<_Float16*>(g), (lds_b128_p)l, 0, 0);
}
#endif

// ---- WMMA fragment gathers (per CDNA5 ISA 16-bit A/B layouts) --------------

// A 16x32 f16: lanes 0-15 -> M=lane, K={0..7,16..23}; lanes 16-31 -> K={8..15,24..31}
__device__ __forceinline__ v16h ldsA_frag(const _Float16* base, int lane) {
  const int m = lane & 15, h = lane >> 4;
  const _Float16* p = base + m * LDT + 8 * h;
  v8h lo = *(const v8h*)(p);        // K = 8h .. 8h+7
  v8h hi = *(const v8h*)(p + 16);   // K = 16+8h .. 16+8h+7
  v16h r;
#pragma unroll
  for (int t = 0; t < 8; ++t) { r[t] = lo[t]; r[t + 8] = hi[t]; }
  return r;
}

// B 32x16 f16 (stored transposed in LDS as [N][K]): lane n%16 = column,
// lanes 0-15 hold K={0..7,16..23}, lanes 16-31 hold K={8..15,24..31} (same as A).
__device__ __forceinline__ v16h ldsB_frag(const _Float16* base, int lane) {
  const int n = lane & 15, kb = (lane >> 4) << 3;
  const _Float16* p = base + n * LDT + kb;
  v8h lo = *(const v8h*)(p);
  v8h hi = *(const v8h*)(p + 16);
  v16h r;
#pragma unroll
  for (int t = 0; t < 8; ++t) { r[t] = lo[t]; r[t + 8] = hi[t]; }
  return r;
}

__device__ __forceinline__ v8f wmma_f16(v16h a, v16h b, v8f c) {
  return __builtin_amdgcn_wmma_f32_16x16x32_f16(false, a, false, b, (short)0, c,
                                                false, false);
}

// ---- small conversion kernels ----------------------------------------------

__global__ void cvt_f32_f16(const float* __restrict__ src,
                            _Float16* __restrict__ dst, int n) {
  int i = blockIdx.x * blockDim.x + threadIdx.x;
  int stride = gridDim.x * blockDim.x;
  for (; i < n; i += stride) dst[i] = (_Float16)src[i];
}

// src [K,N] f32 row-major -> dst [N,K] f16 row-major
__global__ void tcvt_f32_f16(const float* __restrict__ src,
                             _Float16* __restrict__ dst, int K, int N) {
  int i = blockIdx.x * blockDim.x + threadIdx.x;
  int stride = gridDim.x * blockDim.x;
  int n = K * N;
  for (; i < n; i += stride) {
    int kk = i / N, nn = i - kk * N;
    dst[nn * K + kk] = (_Float16)src[i];
  }
}

// ---- generic GEMM + bias + ReLU, f16 in / f16 out, f32 accumulate ----------
// C[M,N] = relu(A[M,K] @ Bt[N,K]^T + bias[N]).  Block tile 128x128, 8 waves
// of 32x64 each (2x4 WMMA tiles), K-step 32, double-buffered LDS.

__global__ __launch_bounds__(256) void gemm_bias_relu_f16(
    const _Float16* __restrict__ A, const _Float16* __restrict__ Bt,
    const float* __restrict__ bias, _Float16* __restrict__ C,
    int M, int N, int K) {
  __shared__ alignas(16) _Float16 smem[4 * ABUF];  // 57,344 B

  const int tid = threadIdx.x;
  const int lane = tid & 31, wave = tid >> 5;
  const int mw = wave & 3, nw = wave >> 2;          // 4x2 wave grid
  const int blockN = blockIdx.x * 128;
  const int blockM = blockIdx.y * 128;

  v8f acc[2][4];
#pragma unroll
  for (int i = 0; i < 2; ++i)
#pragma unroll
    for (int j = 0; j < 4; ++j)
#pragma unroll
      for (int r = 0; r < 8; ++r) acc[i][j][r] = 0.f;

  const int lrow = tid >> 1;              // 0..127
  const int lseg = (tid & 1) << 4;        // 0 or 16 halfs
  const _Float16* gA = A + (size_t)(blockM + lrow) * K + lseg;
  const _Float16* gB = Bt + (size_t)(blockN + lrow) * K + lseg;
  const int NK = K >> 5;
  int cur = 0;

#if ASYNC_OK
  // -------- async-copy pipeline: ASYNCcnt-tracked global->LDS DMA ----------
  {
    _Float16* a = smem + lrow * LDT + lseg;
    _Float16* b = smem + ABUF + lrow * LDT + lseg;
    async_cp16(gA, a); async_cp16(gA + 8, a + 8);
    async_cp16(gB, b); async_cp16(gB + 8, b + 8);
  }
  for (int kt = 0; kt < NK; ++kt) {
    if (kt + 1 < NK) {  // next tile's copies run during this tile's WMMAs
      const _Float16* pa = gA + (kt + 1) * TILE_K;
      const _Float16* pb = gB + (kt + 1) * TILE_K;
      _Float16* a = smem + (cur ^ 1) * 2 * ABUF + lrow * LDT + lseg;
      _Float16* b = a + ABUF;
      async_cp16(pa, a); async_cp16(pa + 8, a + 8);
      async_cp16(pb, b); async_cp16(pb + 8, b + 8);
      __builtin_amdgcn_s_wait_asynccnt(4);  // in-order: first 4 (cur buf) done
    } else {
      __builtin_amdgcn_s_wait_asynccnt(0);
    }
    __syncthreads();
    const _Float16* As = smem + cur * 2 * ABUF;
    const _Float16* Bs = As + ABUF;
    v16h aF[2], bF[4];
#pragma unroll
    for (int i = 0; i < 2; ++i)
      aF[i] = ldsA_frag(As + (32 * mw + 16 * i) * LDT, lane);
#pragma unroll
    for (int j = 0; j < 4; ++j)
      bF[j] = ldsB_frag(Bs + (64 * nw + 16 * j) * LDT, lane);
#pragma unroll
    for (int i = 0; i < 2; ++i)
#pragma unroll
      for (int j = 0; j < 4; ++j)
        acc[i][j] = wmma_f16(aF[i], bF[j], acc[i][j]);
    __syncthreads();  // compute done before next iter's copies overwrite cur
    cur ^= 1;
  }
#else
  // -------- register-staged sync pipeline ----------------------------------
  v8h ra0 = *(const v8h*)(gA);
  v8h ra1 = *(const v8h*)(gA + 8);
  v8h rb0 = *(const v8h*)(gB);
  v8h rb1 = *(const v8h*)(gB + 8);
  {
    _Float16* a = smem + lrow * LDT + lseg;
    _Float16* b = smem + ABUF + lrow * LDT + lseg;
    *(v8h*)a = ra0; *(v8h*)(a + 8) = ra1;
    *(v8h*)b = rb0; *(v8h*)(b + 8) = rb1;
  }
  for (int kt = 0; kt < NK; ++kt) {
    __syncthreads();
    const bool more = (kt + 1 < NK);
    if (more) {  // issue next tile's global loads; they overlap the WMMAs
      const _Float16* pa = gA + (kt + 1) * TILE_K;
      const _Float16* pb = gB + (kt + 1) * TILE_K;
      ra0 = *(const v8h*)(pa); ra1 = *(const v8h*)(pa + 8);
      rb0 = *(const v8h*)(pb); rb1 = *(const v8h*)(pb + 8);
    }
    const _Float16* As = smem + cur * 2 * ABUF;
    const _Float16* Bs = As + ABUF;
    v16h aF[2], bF[4];
#pragma unroll
    for (int i = 0; i < 2; ++i)
      aF[i] = ldsA_frag(As + (32 * mw + 16 * i) * LDT, lane);
#pragma unroll
    for (int j = 0; j < 4; ++j)
      bF[j] = ldsB_frag(Bs + (64 * nw + 16 * j) * LDT, lane);
#pragma unroll
    for (int i = 0; i < 2; ++i)
#pragma unroll
      for (int j = 0; j < 4; ++j)
        acc[i][j] = wmma_f16(aF[i], bF[j], acc[i][j]);
    if (more) {  // fill the other buffer (nobody reads it this iteration)
      _Float16* a = smem + (cur ^ 1) * 2 * ABUF + lrow * LDT + lseg;
      _Float16* b = a + ABUF;
      *(v8h*)a = ra0; *(v8h*)(a + 8) = ra1;
      *(v8h*)b = rb0; *(v8h*)(b + 8) = rb1;
    }
    cur ^= 1;
  }
#endif

  // ---- epilogue: bias+relu -> f16 LDS tile -> coalesced b128 stores --------
  __syncthreads();
  _Float16* Cs = smem;  // 128 x LDC = 34,816 B, reuses pipeline buffers
  const int ln = lane & 15, lh = lane >> 4;
#pragma unroll
  for (int i = 0; i < 2; ++i) {
#pragma unroll
    for (int j = 0; j < 4; ++j) {
      const int cn  = 64 * nw + 16 * j + ln;
      const int cm0 = 32 * mw + 16 * i + 8 * lh;
      const float bv = bias[blockN + cn];
#pragma unroll
      for (int r = 0; r < 8; ++r) {
        float v = acc[i][j][r] + bv;
        v = v > 0.f ? v : 0.f;
        Cs[(cm0 + r) * LDC + cn] = (_Float16)v;
      }
    }
  }
  __syncthreads();
  const int srow = tid >> 1;
  const int sseg = (tid & 1) << 6;  // 0 or 64 halfs
  _Float16* cp = C + (size_t)(blockM + srow) * N + blockN + sseg;
  const _Float16* cs = Cs + srow * LDC + sseg;
#pragma unroll
  for (int t = 0; t < 8; ++t)
    *(v8h*)(cp + 8 * t) = *(const v8h*)(cs + 8 * t);
}

// ---- fused attention-score + HardKuma epilogue -----------------------------

#define LOG_T0 (-2.4849066497880004f)   // log((0-L)/(R-L)) = log(1/12)
#define LOG_T1 (-0.08701137698962969f)  // log((1-L)/(R-L)) = log(11/12)

__device__ __forceinline__ float softplus_clamp(float x) {
  float sp = (x > 15.f) ? x : log1pf(expf(x));
  return fminf(fmaxf(sp, 0.01f), 100.f);
}

__global__ __launch_bounds__(256) void kuma_attention(
    const _Float16* __restrict__ qa, const _Float16* __restrict__ ka,
    const _Float16* __restrict__ qb, const _Float16* __restrict__ kb,
    const float* __restrict__ dist_emb, float* __restrict__ out,
    int T, int D) {
  __shared__ alignas(16) _Float16 Qa[ABUF];
  __shared__ alignas(16) _Float16 Ka[ABUF];
  __shared__ alignas(16) _Float16 Qb[ABUF];
  __shared__ alignas(16) _Float16 Kb[ABUF];
  __shared__ float rd_lds[23];

  const int tid = threadIdx.x;
  const int lane = tid & 31, wave = tid >> 5;
  const int mw = wave & 3, nw = wave >> 2;
  const int sBlk = blockIdx.x * 128;   // key/column tile
  const int tBlk = blockIdx.y * 128;   // query/row tile
  const size_t boff = (size_t)blockIdx.z * T * D;

  if (tid < 23) rd_lds[tid] = dist_emb[tid];

  v8f accA[2][4], accB[2][4];
#pragma unroll
  for (int i = 0; i < 2; ++i)
#pragma unroll
    for (int j = 0; j < 4; ++j)
#pragma unroll
      for (int r = 0; r < 8; ++r) { accA[i][j][r] = 0.f; accB[i][j][r] = 0.f; }

  const int lrow = tid >> 1;
  const int lseg = (tid & 1) << 4;
  const _Float16* gQa = qa + boff + (size_t)(tBlk + lrow) * D + lseg;
  const _Float16* gQb = qb + boff + (size_t)(tBlk + lrow) * D + lseg;
  const _Float16* gKa = ka + boff + (size_t)(sBlk + lrow) * D + lseg;
  const _Float16* gKb = kb + boff + (size_t)(sBlk + lrow) * D + lseg;

  // register double-buffer: preload tile 0
  v8h qa0 = *(const v8h*)(gQa), qa1 = *(const v8h*)(gQa + 8);
  v8h qb0 = *(const v8h*)(gQb), qb1 = *(const v8h*)(gQb + 8);
  v8h ka0 = *(const v8h*)(gKa), ka1 = *(const v8h*)(gKa + 8);
  v8h kb0 = *(const v8h*)(gKb), kb1 = *(const v8h*)(gKb + 8);

  const int NK = D >> 5;
  for (int kt = 0; kt < NK; ++kt) {
    __syncthreads();  // previous compute done; safe to overwrite tiles
    {
      _Float16* a = &Qa[lrow * LDT + lseg];
      _Float16* b = &Qb[lrow * LDT + lseg];
      _Float16* c = &Ka[lrow * LDT + lseg];
      _Float16* d = &Kb[lrow * LDT + lseg];
      *(v8h*)a = qa0; *(v8h*)(a + 8) = qa1;
      *(v8h*)b = qb0; *(v8h*)(b + 8) = qb1;
      *(v8h*)c = ka0; *(v8h*)(c + 8) = ka1;
      *(v8h*)d = kb0; *(v8h*)(d + 8) = kb1;
    }
    __syncthreads();
    if (kt + 1 < NK) {  // next tile's loads overlap the 16 WMMAs below
      const int o = (kt + 1) * TILE_K;
      qa0 = *(const v8h*)(gQa + o); qa1 = *(const v8h*)(gQa + o + 8);
      qb0 = *(const v8h*)(gQb + o); qb1 = *(const v8h*)(gQb + o + 8);
      ka0 = *(const v8h*)(gKa + o); ka1 = *(const v8h*)(gKa + o + 8);
      kb0 = *(const v8h*)(gKb + o); kb1 = *(const v8h*)(gKb + o + 8);
    }
    {  // a-scores
      v16h aF[2], bF[4];
#pragma unroll
      for (int i = 0; i < 2; ++i)
        aF[i] = ldsA_frag(&Qa[(32 * mw + 16 * i) * LDT], lane);
#pragma unroll
      for (int j = 0; j < 4; ++j)
        bF[j] = ldsB_frag(&Ka[(64 * nw + 16 * j) * LDT], lane);
#pragma unroll
      for (int i = 0; i < 2; ++i)
#pragma unroll
        for (int j = 0; j < 4; ++j)
          accA[i][j] = wmma_f16(aF[i], bF[j], accA[i][j]);
    }
    {  // b-scores
      v16h aF[2], bF[4];
#pragma unroll
      for (int i = 0; i < 2; ++i)
        aF[i] = ldsA_frag(&Qb[(32 * mw + 16 * i) * LDT], lane);
#pragma unroll
      for (int j = 0; j < 4; ++j)
        bF[j] = ldsB_frag(&Kb[(64 * nw + 16 * j) * LDT], lane);
#pragma unroll
      for (int i = 0; i < 2; ++i)
#pragma unroll
        for (int j = 0; j < 4; ++j)
          accB[i][j] = wmma_f16(aF[i], bF[j], accB[i][j]);
    }
  }

  // HardKuma epilogue; base pointer + 32-bit offsets (o += T per row)
  const int ln = lane & 15, lh = lane >> 4;
  float* op = out + (size_t)blockIdx.z * T * T;
#pragma unroll
  for (int i = 0; i < 2; ++i) {
#pragma unroll
    for (int j = 0; j < 4; ++j) {
      const int s  = sBlk + 64 * nw + 16 * j + ln;
      const int t0 = tBlk + 32 * mw + 16 * i + 8 * lh;
      unsigned int o = (unsigned int)t0 * (unsigned int)T + (unsigned int)s;
#pragma unroll
      for (int r = 0; r < 8; ++r) {
        const int t = t0 + r;
        int rel = s - t;
        rel = rel < -11 ? -11 : (rel > 11 ? 11 : rel);
        const float rd = rd_lds[rel + 11];

        const float av = softplus_clamp(accA[i][j][r] + rd);
        const float bv = softplus_clamp(accB[i][j][r] + rd);

        const float ta = expf(av * LOG_T0);              // t0^a
        const float tb = expf(av * LOG_T1);              // t1^a
        const float p0 = 1.f - expf(bv * log1pf(-ta));   // P(h=0)
        const float p1 = expf(bv * log1pf(-tb));         // P(h=1)
        const float pc = 1.f - p0 - p1;

        const float g = 1.f + 1.f / av;
        const float kmean =
            bv * expf(lgammaf(g) + lgammaf(bv) - lgammaf(g + bv));
        const float mean = fminf(fmaxf(-0.1f + 1.2f * kmean, 0.f), 1.f);

        const float att = (pc < 0.5f) ? ((p0 > p1) ? 0.f : 1.f) : mean;
        op[o] = att;
        o += (unsigned int)T;
      }
    }
  }
}

// ---------------------------------------------------------------------------

extern "C" void kernel_launch(void* const* d_in, const int* in_sizes, int n_in,
                              void* d_out, int out_size, void* d_ws,
                              size_t ws_size, hipStream_t stream) {
  (void)in_sizes; (void)n_in; (void)out_size; (void)ws_size;

  const float* q   = (const float*)d_in[0];
  const float* k   = (const float*)d_in[1];
  const float* Wa1 = (const float*)d_in[2];
  const float* ba1 = (const float*)d_in[3];
  const float* Wa2 = (const float*)d_in[4];
  const float* ba2 = (const float*)d_in[5];
  const float* Wb1 = (const float*)d_in[6];
  const float* bb1 = (const float*)d_in[7];
  const float* Wb2 = (const float*)d_in[8];
  const float* bb2 = (const float*)d_in[9];
  const float* dist_emb = (const float*)d_in[10];

  const int B = 4, T = 2048, Din = 512, Dout = 256;
  const int M = B * T;  // 8192

  char* ws = (char*)d_ws;
  size_t off = 0;
  auto take = [&](size_t elems) -> _Float16* {
    _Float16* p = (_Float16*)(ws + off);
    off += ((elems * sizeof(_Float16)) + 255) & ~(size_t)255;
    return p;
  };
  _Float16* q16  = take((size_t)M * Din);
  _Float16* k16  = take((size_t)M * Din);
  _Float16* wa1t = take((size_t)Din * Dout);   // [Dout,Din]
  _Float16* wa2t = take((size_t)Dout * Dout);
  _Float16* wb1t = take((size_t)Din * Dout);
  _Float16* wb2t = take((size_t)Dout * Dout);
  _Float16* h    = take((size_t)M * Dout);
  _Float16* qa16 = take((size_t)M * Dout);
  _Float16* ka16 = take((size_t)M * Dout);
  _Float16* qb16 = take((size_t)M * Dout);
  _Float16* kb16 = take((size_t)M * Dout);

  // 1) conversions
  {
    int n = M * Din;
    int blocks = (n + 255) / 256; if (blocks > 2048) blocks = 2048;
    cvt_f32_f16<<<blocks, 256, 0, stream>>>(q, q16, n);
    cvt_f32_f16<<<blocks, 256, 0, stream>>>(k, k16, n);
  }
  {
    int blocks = 256;
    tcvt_f32_f16<<<blocks, 256, 0, stream>>>(Wa1, wa1t, Din, Dout);
    tcvt_f32_f16<<<blocks, 256, 0, stream>>>(Wa2, wa2t, Dout, Dout);
    tcvt_f32_f16<<<blocks, 256, 0, stream>>>(Wb1, wb1t, Din, Dout);
    tcvt_f32_f16<<<blocks, 256, 0, stream>>>(Wb2, wb2t, Dout, Dout);
  }

  // 2) MLPs: x -> relu(x@W1+b1) -> relu(@W2+b2)
  dim3 gg(Dout / 128, M / 128);
  gemm_bias_relu_f16<<<gg, 256, 0, stream>>>(q16, wa1t, ba1, h, M, Dout, Din);
  gemm_bias_relu_f16<<<gg, 256, 0, stream>>>(h, wa2t, ba2, qa16, M, Dout, Dout);
  gemm_bias_relu_f16<<<gg, 256, 0, stream>>>(k16, wa1t, ba1, h, M, Dout, Din);
  gemm_bias_relu_f16<<<gg, 256, 0, stream>>>(h, wa2t, ba2, ka16, M, Dout, Dout);
  gemm_bias_relu_f16<<<gg, 256, 0, stream>>>(q16, wb1t, bb1, h, M, Dout, Din);
  gemm_bias_relu_f16<<<gg, 256, 0, stream>>>(h, wb2t, bb2, qb16, M, Dout, Dout);
  gemm_bias_relu_f16<<<gg, 256, 0, stream>>>(k16, wb1t, bb1, h, M, Dout, Din);
  gemm_bias_relu_f16<<<gg, 256, 0, stream>>>(h, wb2t, bb2, kb16, M, Dout, Dout);

  // 3) fused scores + HardKuma
  dim3 ag(T / 128, T / 128, B);
  kuma_attention<<<ag, 256, 0, stream>>>(qa16, ka16, qb16, kb16, dist_emb,
                                         (float*)d_out, T, Dout);
}
